// MomentTransformer_17016660426756
// MI455X (gfx1250) — compile-verified, matrix operand order checked by reading, not run on hardware
//
#include <hip/hip_runtime.h>
#include <hip/hip_bf16.h>
#include <math.h>

typedef __attribute__((ext_vector_type(16))) __bf16 v16bf;
typedef __attribute__((ext_vector_type(8)))  __bf16 v8bf;
typedef __attribute__((ext_vector_type(4)))  __bf16 v4bf;
typedef __attribute__((ext_vector_type(8)))  float  v8f;

#define EMB   512
#define HEADS 8
#define DH    64
#define FFD   2048
#define BATCH 32
#define SEQ   1024          // 16 channels * 64 timesteps
#define MROWS (BATCH*SEQ)   // 32768

// ---------------------------------------------------------------------------
// Weight transpose + f32 -> bf16 convert: in [R,C] f32 -> out [C,R] bf16
// ---------------------------------------------------------------------------
__global__ void transpose_to_bf16(const float* __restrict__ in,
                                  __bf16* __restrict__ out, int R, int C) {
    int g = blockIdx.x * 256 + threadIdx.x;
    if (g >= R * C) return;
    int r = g / C, c = g % C;
    out[(size_t)c * R + r] = (__bf16)in[g];
}

// ---------------------------------------------------------------------------
// emb[b, ch, t, d] = x[b,t,d] + channel_token[ch,d] + PE[t,d]
// ---------------------------------------------------------------------------
__global__ void build_emb_kernel(const float* __restrict__ x,
                                 const float* __restrict__ ch,
                                 float* __restrict__ emb) {
    int g = blockIdx.x * 256 + threadIdx.x;   // element id, total MROWS*EMB
    int d   = g & 511;
    int row = g >> 9;
    int t   = row & 63;
    int c   = (row >> 6) & 15;
    int b   = row >> 10;
    // PE: freq = exp(-(2i) * ln(10000)/512), even->sin, odd->cos
    float freq = expf(-(float)(d & ~1) * 0.017988946039015984f);
    float ang  = (float)t * freq;
    float pe   = (d & 1) ? cosf(ang) : sinf(ang);
    emb[g] = x[((size_t)b * 64 + t) * EMB + d] + ch[(size_t)c * EMB + d] + pe;
}

// ---------------------------------------------------------------------------
// LayerNorm over d=512, one wave32 per row, output bf16
// ---------------------------------------------------------------------------
__global__ __launch_bounds__(256)
void layernorm_bf16_kernel(const float* __restrict__ x,
                           const float* __restrict__ gam,
                           const float* __restrict__ bet,
                           __bf16* __restrict__ out) {
    int row  = blockIdx.x * 8 + (threadIdx.x >> 5);
    int lane = threadIdx.x & 31;
    const float4* xr = (const float4*)(x + (size_t)row * EMB);
    const float4* g4 = (const float4*)gam;
    const float4* b4 = (const float4*)bet;
    float4 v[4];
    float s = 0.f, s2 = 0.f;
#pragma unroll
    for (int i = 0; i < 4; ++i) {
        v[i] = xr[i * 32 + lane];
        s  += v[i].x + v[i].y + v[i].z + v[i].w;
        s2 += v[i].x*v[i].x + v[i].y*v[i].y + v[i].z*v[i].z + v[i].w*v[i].w;
    }
#pragma unroll
    for (int m = 16; m >= 1; m >>= 1) {
        s  += __shfl_xor(s,  m);
        s2 += __shfl_xor(s2, m);
    }
    float mean = s * (1.0f / EMB);
    float var  = s2 * (1.0f / EMB) - mean * mean;
    float inv  = rsqrtf(var + 1e-5f);
#pragma unroll
    for (int i = 0; i < 4; ++i) {
        float4 gg = g4[i * 32 + lane];
        float4 bb = b4[i * 32 + lane];
        v4bf o;
        o[0] = (__bf16)((v[i].x - mean) * inv * gg.x + bb.x);
        o[1] = (__bf16)((v[i].y - mean) * inv * gg.y + bb.y);
        o[2] = (__bf16)((v[i].z - mean) * inv * gg.z + bb.z);
        o[3] = (__bf16)((v[i].w - mean) * inv * gg.w + bb.w);
        *(v4bf*)(out + (size_t)row * EMB + (size_t)(i * 32 + lane) * 4) = o;
    }
}

// ---------------------------------------------------------------------------
// bf16 WMMA GEMM:  C[M,N] = A[M,K] * Bt[N,K]^T   (Bt row-major = B col-major)
// Wave tile 32x64 (2x4 wmma 16x16x32 tiles), 8 waves/block -> 64x256 per block
// mode 0: Cf = acc
// mode 1: Cf = resid + acc + bias[col]          (residual update, may alias)
// mode 2: Cb = bf16( gelu(acc + bias[col]) )    (exact-erf GELU)
// ---------------------------------------------------------------------------
__device__ __forceinline__ v16bf load_frag(const __bf16* p) {
    v8bf lo = *(const v8bf*)(p);
    v8bf hi = *(const v8bf*)(p + 16);
    v16bf r;
#pragma unroll
    for (int i = 0; i < 8; ++i) { r[i] = lo[i]; r[i + 8] = hi[i]; }
    return r;
}

__global__ __launch_bounds__(256)
void gemm_bf16_wmma(const __bf16* __restrict__ A, const __bf16* __restrict__ Bt,
                    float* __restrict__ Cf, __bf16* __restrict__ Cb,
                    const float* __restrict__ bias, const float* __restrict__ resid,
                    int M, int N, int K, int mode) {
    const int lane = threadIdx.x & 31;
    const int wid  = threadIdx.x >> 5;
    const int wm = wid >> 2, wn = wid & 3;
    const int rowBase = blockIdx.y * 64 + wm * 32;
    const int colBase = blockIdx.x * 256 + wn * 64;
    const int lrow = lane & 15;
    const int lk   = (lane >> 4) * 8;

    v8f acc[2][4] = {};

    const __bf16* a0p = A + (size_t)(rowBase + lrow) * K + lk;
    const __bf16* a1p = A + (size_t)(rowBase + 16 + lrow) * K + lk;
    const __bf16* bp0 = Bt + (size_t)(colBase +  0 + lrow) * K + lk;
    const __bf16* bp1 = Bt + (size_t)(colBase + 16 + lrow) * K + lk;
    const __bf16* bp2 = Bt + (size_t)(colBase + 32 + lrow) * K + lk;
    const __bf16* bp3 = Bt + (size_t)(colBase + 48 + lrow) * K + lk;

    for (int k0 = 0; k0 < K; k0 += 32) {
        v16bf a0 = load_frag(a0p + k0);
        v16bf a1 = load_frag(a1p + k0);
        v16bf b0 = load_frag(bp0 + k0);
        v16bf b1 = load_frag(bp1 + k0);
        v16bf b2 = load_frag(bp2 + k0);
        v16bf b3 = load_frag(bp3 + k0);
        acc[0][0] = __builtin_amdgcn_wmma_f32_16x16x32_bf16(false, a0, false, b0, (short)0, acc[0][0], false, false);
        acc[0][1] = __builtin_amdgcn_wmma_f32_16x16x32_bf16(false, a0, false, b1, (short)0, acc[0][1], false, false);
        acc[0][2] = __builtin_amdgcn_wmma_f32_16x16x32_bf16(false, a0, false, b2, (short)0, acc[0][2], false, false);
        acc[0][3] = __builtin_amdgcn_wmma_f32_16x16x32_bf16(false, a0, false, b3, (short)0, acc[0][3], false, false);
        acc[1][0] = __builtin_amdgcn_wmma_f32_16x16x32_bf16(false, a1, false, b0, (short)0, acc[1][0], false, false);
        acc[1][1] = __builtin_amdgcn_wmma_f32_16x16x32_bf16(false, a1, false, b1, (short)0, acc[1][1], false, false);
        acc[1][2] = __builtin_amdgcn_wmma_f32_16x16x32_bf16(false, a1, false, b2, (short)0, acc[1][2], false, false);
        acc[1][3] = __builtin_amdgcn_wmma_f32_16x16x32_bf16(false, a1, false, b3, (short)0, acc[1][3], false, false);
    }

#pragma unroll
    for (int i = 0; i < 2; ++i) {
#pragma unroll
        for (int j = 0; j < 4; ++j) {
            int col   = colBase + j * 16 + lrow;
            int rbase = rowBase + i * 16 + lk;   // C: M = (lane>>4)*8 + r
            float bcol = (mode != 0) ? bias[col] : 0.f;
#pragma unroll
            for (int r = 0; r < 8; ++r) {
                size_t off = (size_t)(rbase + r) * N + col;
                float val = acc[i][j][r];
                if (mode == 0) {
                    Cf[off] = val;
                } else if (mode == 1) {
                    Cf[off] = resid[off] + val + bcol;
                } else {
                    float t = val + bcol;
                    float ge = 0.5f * t * (1.0f + erff(t * 0.70710678118654752f));
                    Cb[off] = (__bf16)ge;
                }
            }
        }
    }
}

// ---------------------------------------------------------------------------
// softmax over dh=64 (last dim per head) then * DH^-0.5; one wave per (row,head)
// ---------------------------------------------------------------------------
__global__ __launch_bounds__(256)
void softmax_q_kernel(float* __restrict__ q) {
    int rh   = blockIdx.x * 8 + (threadIdx.x >> 5);   // m*8 + h
    int lane = threadIdx.x & 31;
    float* p = q + (size_t)(rh >> 3) * EMB + (size_t)(rh & 7) * DH;
    float v0 = p[lane], v1 = p[lane + 32];
    float mx = fmaxf(v0, v1);
#pragma unroll
    for (int m = 16; m >= 1; m >>= 1) mx = fmaxf(mx, __shfl_xor(mx, m));
    float e0 = expf(v0 - mx), e1 = expf(v1 - mx);
    float s = e0 + e1;
#pragma unroll
    for (int m = 16; m >= 1; m >>= 1) s += __shfl_xor(s, m);
    float inv = 0.125f / s;   // DH^-0.5 = 1/8
    p[lane] = e0 * inv;
    p[lane + 32] = e1 * inv;
}

// ---------------------------------------------------------------------------
// softmax over sequence n=1024 (per batch, per column); one thread per column
// ---------------------------------------------------------------------------
__global__ void softmax_k_kernel(float* __restrict__ k) {
    int g = blockIdx.x * 256 + threadIdx.x;   // 0 .. 32*512-1
    int b = g >> 9, col = g & 511;
    float* base = k + (size_t)b * SEQ * EMB + col;
    float mx = -INFINITY, s = 0.f;
    for (int r = 0; r < SEQ; ++r) {
        float v = base[(size_t)r * EMB];
        float nm = fmaxf(mx, v);
        s = s * expf(mx - nm) + expf(v - nm);
        mx = nm;
    }
    float invs = 1.0f / s;
    for (int r = 0; r < SEQ; ++r) {
        float v = base[(size_t)r * EMB];
        base[(size_t)r * EMB] = expf(v - mx) * invs;
    }
}

// ---------------------------------------------------------------------------
// ctx[b,h,d,e] = sum_n k[b,h,n,d] * v[b,h,n,e]   (64x64x1024 per head)
// ---------------------------------------------------------------------------
__global__ __launch_bounds__(256)
void context_kernel(const float* __restrict__ k, const float* __restrict__ v,
                    float* __restrict__ ctx) {
    int bh = blockIdx.x;             // b*8 + h
    int b = bh >> 3, h = bh & 7;
    __shared__ float ks[8][64];
    __shared__ float vs[8][64];
    int t = threadIdx.x;
    int d = t & 63, grp = t >> 6;
    float acc[16] = {};
    for (int n0 = 0; n0 < SEQ; n0 += 8) {
        for (int idx = t; idx < 512; idx += 256) {
            int r = idx >> 6, c = idx & 63;
            size_t off = ((size_t)(b * SEQ + n0 + r)) * EMB + h * DH + c;
            ks[r][c] = k[off];
            vs[r][c] = v[off];
        }
        __syncthreads();
#pragma unroll
        for (int j = 0; j < 8; ++j) {
            float kv = ks[j][d];
#pragma unroll
            for (int e = 0; e < 16; ++e) acc[e] += kv * vs[j][grp * 16 + e];
        }
        __syncthreads();
    }
    float* o = ctx + (size_t)bh * 4096 + (size_t)d * 64 + grp * 16;
#pragma unroll
    for (int e = 0; e < 16; ++e) o[e] = acc[e];
}

// ---------------------------------------------------------------------------
// attn[m, h*64+e] = sum_d q[m, h*64+d] * ctx[b,h,d,e]  -> bf16 (input to Wo)
// block = 64 rows of one (b,h)
// ---------------------------------------------------------------------------
__global__ __launch_bounds__(256)
void attn_kernel(const float* __restrict__ q, const float* __restrict__ ctx,
                 __bf16* __restrict__ attn) {
    int blk = blockIdx.x;            // b*128 + h*16 + rowchunk
    int rc = blk & 15;
    int h  = (blk >> 4) & 7;
    int b  = blk >> 7;
    __shared__ float qs[64][65];
    __shared__ float cs[64][65];
    int t = threadIdx.x;
    int rowbase = b * SEQ + rc * 64;
    for (int idx = t; idx < 4096; idx += 256) {
        int r = idx >> 6, c = idx & 63;
        qs[r][c] = q[((size_t)(rowbase + r)) * EMB + h * DH + c];
        cs[r][c] = ctx[((size_t)(b * 8 + h)) * 4096 + idx];
    }
    __syncthreads();
    int ml = t >> 2, g = t & 3, e0 = g * 16;
    float acc[16] = {};
#pragma unroll 8
    for (int d = 0; d < 64; ++d) {
        float qv = qs[ml][d];
#pragma unroll
        for (int e = 0; e < 16; ++e) acc[e] += qv * cs[d][e0 + e];
    }
    __bf16* o = attn + ((size_t)(rowbase + ml)) * EMB + h * DH + e0;
#pragma unroll
    for (int e = 0; e < 16; ++e) o[e] = (__bf16)acc[e];
}

// ---------------------------------------------------------------------------
// out[b,d] = mean over n of emb[b*1024+n, d]
// ---------------------------------------------------------------------------
__global__ void mean_kernel(const float* __restrict__ emb, float* __restrict__ out) {
    int g = blockIdx.x * 256 + threadIdx.x;   // 0 .. 32*512-1
    int b = g >> 9, d = g & 511;
    const float* p = emb + (size_t)b * SEQ * EMB + d;
    float s = 0.f;
    for (int r = 0; r < SEQ; ++r) s += p[(size_t)r * EMB];
    out[g] = s * (1.0f / SEQ);
}

// ---------------------------------------------------------------------------
extern "C" void kernel_launch(void* const* d_in, const int* in_sizes, int n_in,
                              void* d_out, int out_size, void* d_ws, size_t ws_size,
                              hipStream_t stream) {
    const float* x    = (const float*)d_in[0];
    const float* chtk = (const float*)d_in[1];
    const float* ln1g = (const float*)d_in[2];
    const float* ln1b = (const float*)d_in[3];
    const float* wq   = (const float*)d_in[4];
    const float* wk   = (const float*)d_in[5];
    const float* wv   = (const float*)d_in[6];
    const float* wo   = (const float*)d_in[7];
    const float* wob  = (const float*)d_in[8];
    const float* ln2g = (const float*)d_in[9];
    const float* ln2b = (const float*)d_in[10];
    const float* w1   = (const float*)d_in[11];
    const float* b1   = (const float*)d_in[12];
    const float* w2   = (const float*)d_in[13];
    const float* b2   = (const float*)d_in[14];

    char* ws = (char*)d_ws;
    float*  emb   = (float*) (ws);                         // 67108864 B
    __bf16* hbuf  = (__bf16*)(ws + 67108864);              // 33554432 B
    float*  qbuf  = (float*) (ws + 100663296);             // 67108864 B
    float*  kbuf  = (float*) (ws + 167772160);             // 67108864 B
    float*  vbuf  = (float*) (ws + 234881024);             // 67108864 B
    __bf16* attnb = (__bf16*)(ws + 301989888);             // 33554432 B
    float*  ctx   = (float*) (ws + 335544320);             //  4194304 B
    __bf16* ffb   = (__bf16*)(ws + 339738624);             // 134217728 B
    __bf16* wqT   = (__bf16*)(ws + 473956352);
    __bf16* wkT   = wqT + 512 * 512;
    __bf16* wvT   = wkT + 512 * 512;
    __bf16* woT   = wvT + 512 * 512;
    __bf16* w1T   = woT + 512 * 512;        // [2048,512]
    __bf16* w2T   = w1T + 2048 * 512;       // [512,2048]

    // weight prep (tiny)
    transpose_to_bf16<<<1024, 256, 0, stream>>>(wq, wqT, 512, 512);
    transpose_to_bf16<<<1024, 256, 0, stream>>>(wk, wkT, 512, 512);
    transpose_to_bf16<<<1024, 256, 0, stream>>>(wv, wvT, 512, 512);
    transpose_to_bf16<<<1024, 256, 0, stream>>>(wo, woT, 512, 512);
    transpose_to_bf16<<<4096, 256, 0, stream>>>(w1, w1T, 512, 2048);
    transpose_to_bf16<<<4096, 256, 0, stream>>>(w2, w2T, 2048, 512);

    // embedding + PE
    build_emb_kernel<<<(MROWS * EMB) / 256, 256, 0, stream>>>(x, chtk, emb);

    // attention block
    layernorm_bf16_kernel<<<MROWS / 8, 256, 0, stream>>>(emb, ln1g, ln1b, hbuf);
    dim3 gD(EMB / 256, MROWS / 64);     // (2, 512)
    gemm_bf16_wmma<<<gD, 256, 0, stream>>>(hbuf, wqT, qbuf, nullptr, nullptr, nullptr, MROWS, EMB, EMB, 0);
    gemm_bf16_wmma<<<gD, 256, 0, stream>>>(hbuf, wkT, kbuf, nullptr, nullptr, nullptr, MROWS, EMB, EMB, 0);
    gemm_bf16_wmma<<<gD, 256, 0, stream>>>(hbuf, wvT, vbuf, nullptr, nullptr, nullptr, MROWS, EMB, EMB, 0);
    softmax_q_kernel<<<(MROWS * HEADS) / 8, 256, 0, stream>>>(qbuf);
    softmax_k_kernel<<<(BATCH * EMB) / 256, 256, 0, stream>>>(kbuf);
    context_kernel<<<BATCH * HEADS, 256, 0, stream>>>(kbuf, vbuf, ctx);
    attn_kernel<<<BATCH * HEADS * 16, 256, 0, stream>>>(qbuf, ctx, attnb);
    gemm_bf16_wmma<<<gD, 256, 0, stream>>>(attnb, woT, emb, nullptr, wob, emb, MROWS, EMB, EMB, 1);

    // feedforward block
    layernorm_bf16_kernel<<<MROWS / 8, 256, 0, stream>>>(emb, ln2g, ln2b, hbuf);
    dim3 gF(FFD / 256, MROWS / 64);     // (8, 512)
    gemm_bf16_wmma<<<gF, 256, 0, stream>>>(hbuf, w1T, nullptr, ffb, b1, nullptr, MROWS, FFD, EMB, 2);
    gemm_bf16_wmma<<<gD, 256, 0, stream>>>(ffb, w2T, emb, nullptr, b2, emb, MROWS, EMB, FFD, 1);

    // mean over sequence
    mean_kernel<<<(BATCH * EMB) / 256, 256, 0, stream>>>(emb, (float*)d_out);
}
